// SelfAttentionV3_26860725469572
// MI455X (gfx1250) — matrix operand order repaired and checked by reading, not compiled, not run on hardware
//
#include <hip/hip_runtime.h>

typedef __attribute__((ext_vector_type(16))) _Float16 v16h;
typedef __attribute__((ext_vector_type(8)))  float    v8f;
typedef __attribute__((ext_vector_type(4)))  int      v4i;

#define DIM   256
#define BATCH 4
#define SEQ   4096
#define NROWS (BATCH*SEQ)      // 16384 total (b,s) rows
#define SCALE 0.0625f          // 1/sqrt(256)

#define VT_STRIDE  264         // 256 + 8 halfs pad (row-major V tile)
#define VTT_STRIDE 40          // 32 + 8 halfs pad (transposed V tile)

#if defined(__has_builtin)
#  if __has_builtin(__builtin_amdgcn_global_load_async_to_lds_b128)
#    define HAVE_ASYNC_LDS 1
#  endif
#endif
#ifndef HAVE_ASYNC_LDS
#  define HAVE_ASYNC_LDS 0
#endif

// Load a 16-half WMMA fragment as two 16B chunks (global or LDS).
__device__ __forceinline__ v16h ld_frag16(const _Float16* p0, const _Float16* p1) {
    union { v16h h; float4 f[2]; } u;
    u.f[0] = *reinterpret_cast<const float4*>(p0);
    u.f[1] = *reinterpret_cast<const float4*>(p1);
    return u.h;
}

// 16B global -> LDS copy. Async (VGPR-bypassing, ASYNCcnt-tracked) when the
// gfx1250 builtin is available; plain copy otherwise.
__device__ __forceinline__ void stage16(const _Float16* g, _Float16* l) {
#if HAVE_ASYNC_LDS
    typedef __attribute__((address_space(1))) v4i gv4i;   // global int4
    typedef __attribute__((address_space(3))) v4i lv4i;   // LDS int4
    __builtin_amdgcn_global_load_async_to_lds_b128(
        (gv4i*)(unsigned long long)(const void*)g,
        (lv4i*)(unsigned)(unsigned long long)(const void*)l,
        0, 0);
#else
    *reinterpret_cast<float4*>(l) = *reinterpret_cast<const float4*>(g);
#endif
}

// ---------------------------------------------------------------------------
// Kernel 0: convert x (f32) and W (f32) to f16 staging buffers in workspace.
// ---------------------------------------------------------------------------
__global__ void cvt_f16_kernel(const float* __restrict__ x, const float* __restrict__ W,
                               _Float16* __restrict__ x16, _Float16* __restrict__ w16,
                               int nx, int nw) {
    int i = blockIdx.x * blockDim.x + threadIdx.x;
    if (i < nx)            x16[i]      = (_Float16)x[i];
    else if (i < nx + nw)  w16[i - nx] = (_Float16)W[i - nx];
}

// ---------------------------------------------------------------------------
// Kernel 1: QKV projection via WMMA. Only q (e in [0,256)) and v (e in
// [512,768)) are ever used by the reference, so k's third is skipped.
// One wave computes one 16x16 output tile; K = 256 -> 8 WMMAs.
// ---------------------------------------------------------------------------
__global__ void __launch_bounds__(128)
qkv_wmma_kernel(const _Float16* __restrict__ x16, const _Float16* __restrict__ w16,
                const float* __restrict__ bias,
                _Float16* __restrict__ q16, _Float16* __restrict__ v16) {
    const int lane = threadIdx.x & 31;
    const int wave = threadIdx.x >> 5;
    const int n    = lane & 15;
    const int hi   = lane >> 4;                  // lane half: 0 or 1
    const int rowTile = blockIdx.x >> 3;
    const int colTile = (blockIdx.x & 7) * 4 + wave;   // 0..31
    const int r0 = rowTile * 16;

    int e0; _Float16* dst; int dcol0;
    if (colTile < 16) { e0 = colTile * 16;              dst = q16; dcol0 = e0; }
    else              { e0 = 512 + (colTile - 16) * 16; dst = v16; dcol0 = e0 - 512; }

    v8f c = {0.f,0.f,0.f,0.f,0.f,0.f,0.f,0.f};
#pragma unroll
    for (int kk = 0; kk < 8; ++kk) {
        // A = x tile [16 x 32], lane = row m, K contiguous per lane half
        const _Float16* ap = x16 + (size_t)(r0 + n) * DIM + kk * 32 + hi * 16;
        v16h a = ld_frag16(ap, ap + 8);
        // B = W^T tile [32 x 16]: B[k=d][n=e] = W[e][d] -> contiguous per lane
        const _Float16* bp = w16 + (size_t)(e0 + n) * DIM + kk * 32 + hi * 16;
        v16h b = ld_frag16(bp, bp + 8);
        c = __builtin_amdgcn_wmma_f32_16x16x32_f16(false, a, false, b, (short)0, c, false, false);
    }
    const float bb = bias[e0 + n];
#pragma unroll
    for (int r = 0; r < 8; ++r) {
        const int row = r0 + r + hi * 8;         // C layout: VGPR r -> M=r / r+8
        dst[(size_t)row * DIM + dcol0 + n] = (_Float16)(c[r] + bb);
    }
}

// ---------------------------------------------------------------------------
// Kernel 2: flash attention. scores = q@v^T * 1/16, mask cols, online
// softmax, out = P@v. One wave owns a 16-query tile; 4 waves per block share
// each 32-key V tile via double-buffered LDS filled by async global->LDS
// copies that overlap with the WMMA compute of the previous tile.
// Grid: BATCH*(SEQ/16)/4 = 256 blocks of 128 threads.
// ---------------------------------------------------------------------------
__global__ void __launch_bounds__(128)
flash_attn_kernel(const _Float16* __restrict__ q16, const _Float16* __restrict__ v16,
                  const int* __restrict__ mask, float* __restrict__ out) {
    __shared__ _Float16 v_tile[2][32 * VT_STRIDE];  // [buf][key][d]
    __shared__ _Float16 v_tileT[DIM * VTT_STRIDE];  // [d][key]
    __shared__ _Float16 pbuf[4][16 * 32];           // per-wave P transpose scratch

    const int tid  = threadIdx.x;
    const int lane = tid & 31;
    const int wave = tid >> 5;
    const int n    = lane & 15;
    const int hi   = lane >> 4;

    const int batch       = blockIdx.x >> 6;                 // 64 blocks / batch
    const int tileInBatch = (blockIdx.x & 63) * 4 + wave;
    const int s0          = tileInBatch * 16;
    const size_t rowBase  = (size_t)batch * SEQ + s0;

    const _Float16* vbase = v16 + (size_t)batch * SEQ * DIM;
    const int* mbase = mask + (size_t)batch * SEQ;
    _Float16* pw = pbuf[wave];

    // Issue one 32x256-half V-tile batch: 1024 16B chunks, 8 per thread,
    // consecutive lanes -> consecutive 16B (coalesced). 8 async ops / wave.
    auto issue_batch = [&](int key0, _Float16* dstTile) {
#pragma unroll
        for (int i = 0; i < 8; ++i) {
            const int ch  = i * 128 + tid;
            const int row = ch >> 5;          // 32 chunks per 256-half row
            const int col = (ch & 31) * 8;
            stage16(vbase + (size_t)(key0 + row) * DIM + col,
                    dstTile + row * VT_STRIDE + col);
        }
    };

    // Q fragments: A-operand, 16 rows x K=256 (8 frags), kept in VGPRs.
    v16h qf[8];
#pragma unroll
    for (int kk = 0; kk < 8; ++kk) {
        const _Float16* qp = q16 + (rowBase + n) * DIM + kk * 32 + hi * 16;
        qf[kk] = ld_frag16(qp, qp + 8);
    }

    const v8f vzero = {0.f,0.f,0.f,0.f,0.f,0.f,0.f,0.f};
    v8f acc[16];
#pragma unroll
    for (int i = 0; i < 16; ++i) acc[i] = vzero;
    const float NEGINF = -__builtin_huge_valf();
    float mrow[8], lrow[8];
#pragma unroll
    for (int r = 0; r < 8; ++r) { mrow[r] = NEGINF; lrow[r] = 0.f; }

    // Software pipeline prologue: batches 0 and 1 in flight.
    issue_batch(0,  v_tile[0]);
    issue_batch(32, v_tile[1]);

    for (int kp = 0; kp < SEQ / 32; ++kp) {
        const int key0 = kp * 32;
        const int cur  = kp & 1;

        // Batch kp complete (async loads complete in order): allow the next
        // batch's 8 ops to stay in flight, except on the final iteration.
        if (kp + 1 < SEQ / 32) {
            asm volatile("s_wait_asynccnt 0x8" ::: "memory");
        } else {
            asm volatile("s_wait_asynccnt 0x0" ::: "memory");
        }
        __syncthreads();

        // Build transposed copy for the P@V B-operand (contiguous key reads).
        {
            const int flat = tid * 64;
            const int row  = flat >> 8;
            const int col  = flat & 255;
            const _Float16* src = &v_tile[cur][row * VT_STRIDE + col];
#pragma unroll
            for (int j = 0; j < 8; ++j) {
                union { float4 f; _Float16 h[8]; } u;
                u.f = *reinterpret_cast<const float4*>(src + j * 8);
#pragma unroll
                for (int i = 0; i < 8; ++i)
                    v_tileT[(col + j * 8 + i) * VTT_STRIDE + row] = u.h[i];
            }
        }
        __syncthreads();

        // Scores for the two 16-key tiles of this 32-key block.
        v8f c0 = vzero, c1 = vzero;
#pragma unroll
        for (int kk = 0; kk < 8; ++kk) {
            const _Float16* b0p = &v_tile[cur][n * VT_STRIDE + kk * 32 + hi * 16];
            v16h b0 = ld_frag16(b0p, b0p + 8);
            c0 = __builtin_amdgcn_wmma_f32_16x16x32_f16(false, qf[kk], false, b0, (short)0, c0, false, false);
            const _Float16* b1p = &v_tile[cur][(n + 16) * VT_STRIDE + kk * 32 + hi * 16];
            v16h b1 = ld_frag16(b1p, b1p + 8);
            c1 = __builtin_amdgcn_wmma_f32_16x16x32_f16(false, qf[kk], false, b1, (short)0, c1, false, false);
        }

        const float madd0 = mbase[key0 + n]      ? 0.f : NEGINF;
        const float madd1 = mbase[key0 + 16 + n] ? 0.f : NEGINF;

        // Online softmax (row r spread over 16 lanes -> butterfly reductions).
#pragma unroll
        for (int r = 0; r < 8; ++r) {
            float s0v = c0[r] * SCALE + madd0;
            float s1v = c1[r] * SCALE + madd1;
            float rmax = fmaxf(s0v, s1v);
            rmax = fmaxf(rmax, __shfl_xor(rmax, 1));
            rmax = fmaxf(rmax, __shfl_xor(rmax, 2));
            rmax = fmaxf(rmax, __shfl_xor(rmax, 4));
            rmax = fmaxf(rmax, __shfl_xor(rmax, 8));
            const float mnew = fmaxf(mrow[r], rmax);
            const float mref = (mnew == NEGINF) ? 0.f : mnew;  // avoid inf-inf NaN
            const float p0 = __expf(s0v - mref);
            const float p1 = __expf(s1v - mref);
            const float alpha = __expf(mrow[r] - mref);
            float psum = p0 + p1;
            psum += __shfl_xor(psum, 1);
            psum += __shfl_xor(psum, 2);
            psum += __shfl_xor(psum, 4);
            psum += __shfl_xor(psum, 8);
            lrow[r] = lrow[r] * alpha + psum;
            mrow[r] = mnew;
#pragma unroll
            for (int dt = 0; dt < 16; ++dt) acc[dt][r] *= alpha;
            const int rl = r + hi * 8;                     // C-layout row
            pw[rl * 32 + n]      = (_Float16)p0;
            pw[rl * 32 + 16 + n] = (_Float16)p1;
        }

        // P (16x32, A-layout from per-wave LDS scratch) @ V (32x256).
        const _Float16* pap = &pw[n * 32 + hi * 16];
        v16h pa = ld_frag16(pap, pap + 8);
#pragma unroll
        for (int dt = 0; dt < 16; ++dt) {
            const _Float16* bp = &v_tileT[(dt * 16 + n) * VTT_STRIDE + hi * 16];
            v16h bv = ld_frag16(bp, bp + 8);
            acc[dt] = __builtin_amdgcn_wmma_f32_16x16x32_f16(false, pa, false, bv, (short)0, acc[dt], false, false);
        }

        // All waves done reading v_tile[cur]; refill it for batch kp+2.
        __syncthreads();
        if (kp + 2 < SEQ / 32) issue_batch(key0 + 64, v_tile[cur]);
    }

    // Epilogue: out = acc / l
#pragma unroll
    for (int dt = 0; dt < 16; ++dt) {
#pragma unroll
        for (int r = 0; r < 8; ++r) {
            const size_t row = rowBase + r + hi * 8;
            out[row * DIM + dt * 16 + n] = acc[dt][r] / lrow[r];
        }
    }
}

// ---------------------------------------------------------------------------
extern "C" void kernel_launch(void* const* d_in, const int* in_sizes, int n_in,
                              void* d_out, int out_size, void* d_ws, size_t ws_size,
                              hipStream_t stream) {
    (void)in_sizes; (void)n_in; (void)out_size; (void)ws_size;
    const float* x    = (const float*)d_in[0];
    const float* W    = (const float*)d_in[1];
    const float* bias = (const float*)d_in[2];
    const int*   mask = (const int*)d_in[3];
    float* out = (float*)d_out;

    _Float16* ws = (_Float16*)d_ws;
    const size_t nx = (size_t)NROWS * DIM;   // 4,194,304 halfs
    const size_t nw = (size_t)768 * DIM;     //   196,608 halfs
    _Float16* x16 = ws;
    _Float16* w16 = x16 + nx;
    _Float16* q16 = w16 + nw;
    _Float16* v16 = q16 + nx;

    {
        const int total  = (int)(nx + nw);
        const int blocks = (total + 255) / 256;
        cvt_f16_kernel<<<blocks, 256, 0, stream>>>(x, W, x16, w16, (int)nx, (int)nw);
    }
    qkv_wmma_kernel<<<(NROWS / 16) * 8, 128, 0, stream>>>(x16, w16, bias, q16, v16);
    flash_attn_kernel<<<(BATCH * (SEQ / 16)) / 4, 128, 0, stream>>>(q16, v16, mask, out);
}
